// FeatureAdaption_30236569764194
// MI455X (gfx1250) — matrix operand order repaired and checked by reading, not compiled, not run on hardware
//
#include <hip/hip_runtime.h>
#include <math.h>

// ---------------- problem constants (from reference) ----------------
constexpr int BATCH = 8;
constexpr int CIN   = 256;
constexpr int HH    = 64;
constexpr int WW    = 64;
constexpr int HWN   = HH * WW;          // 4096
constexpr int DG    = 4;
constexpr int CG    = CIN / DG;         // 64
constexpr int KPTS  = 9;                // 3x3
constexpr int CK    = CIN * KPTS;       // 2304 (GEMM K)
constexpr int GK    = CG * KPTS;        // 576 K values per deform group
constexpr int COUT  = 256;
constexpr int PCH   = 34;               // pred_init channels

// ---------------- GEMM tiling ----------------
constexpr int M_TILE = 256;             // full Cout per block (max gather reuse)
constexpr int N_TILE = 32;              // pixels per block
constexpr int K_STEP = 32;              // WMMA K per step (f16)
constexpr int LDS_STRIDE = 48;          // halves; 96 B rows keep 32 B alignment

typedef __attribute__((ext_vector_type(16))) _Float16 v16h;
typedef __attribute__((ext_vector_type(8)))  float    v8f;
typedef __attribute__((ext_vector_type(8)))  _Float16 h8;
typedef __attribute__((ext_vector_type(4)))  _Float16 h4;

struct __align__(16) Tab {
    int4   idx;   // 4 clamped corner indices into a 64x64 plane
    float4 w;     // 4 bilinear weights (zeroed when corner OOB)
};

// =====================================================================
// Kernel 1: offsets (1x1 conv) + bilinear sampling table.
// One thread per (b, g, k, pixel).  gid = ((b*DG+g)*KPTS+k)*HWN + pix
// =====================================================================
__global__ __launch_bounds__(256)
void dcn_precomp_kernel(const float* __restrict__ pred,
                        const float* __restrict__ w_off,   // [72, 34]
                        Tab* __restrict__ tab)
{
    const int gid = blockIdx.x * blockDim.x + threadIdx.x;
    const int NT  = BATCH * DG * KPTS * HWN;
    if (gid >= NT) return;

    const int pix = gid & (HWN - 1);
    int t = gid >> 12;                 // / 4096
    const int k = t % KPTS; t /= KPTS;
    const int g = t & 3;
    const int b = t >> 2;
    const int h = pix >> 6, w = pix & 63;

    // offset channel o = ((g*9 + k)*2 + {0:y, 1:x})
    const int row_y = (g * KPTS + k) << 1;
    const float* pr = pred + (size_t)b * PCH * HWN + pix;
    const float* wy = w_off + (size_t)row_y * PCH;
    const float* wx = wy + PCH;

    float oy = 0.f, ox = 0.f;
    #pragma unroll
    for (int c = 0; c < PCH; ++c) {
        const float p = pr[(size_t)c * HWN];
        oy = fmaf(p, wy[c], oy);
        ox = fmaf(p, wx[c], ox);
    }

    const float py = (float)(h + (k / 3) - 1) + oy;
    const float px = (float)(w + (k % 3) - 1) + ox;

    const float y0f = floorf(py), x0f = floorf(px);
    const float ty = py - y0f,   tx = px - x0f;

    const float vy0 = (y0f >= 0.f && y0f <= 63.f) ? 1.f : 0.f;
    const float vy1 = (y0f + 1.f >= 0.f && y0f + 1.f <= 63.f) ? 1.f : 0.f;
    const float vx0 = (x0f >= 0.f && x0f <= 63.f) ? 1.f : 0.f;
    const float vx1 = (x0f + 1.f >= 0.f && x0f + 1.f <= 63.f) ? 1.f : 0.f;

    const int y0c = (int)fminf(fmaxf(y0f, 0.f), 63.f);
    const int y1c = (int)fminf(fmaxf(y0f + 1.f, 0.f), 63.f);
    const int x0c = (int)fminf(fmaxf(x0f, 0.f), 63.f);
    const int x1c = (int)fminf(fmaxf(x0f + 1.f, 0.f), 63.f);

    Tab e;
    e.idx = make_int4(y0c * WW + x0c, y0c * WW + x1c,
                      y1c * WW + x0c, y1c * WW + x1c);
    e.w = make_float4((1.f - ty) * (1.f - tx) * vy0 * vx0,
                      (1.f - ty) * tx         * vy0 * vx1,
                      ty * (1.f - tx)         * vy1 * vx0,
                      ty * tx                 * vy1 * vx1);
    tab[gid] = e;
}

// =====================================================================
// Kernel 2: fused deformable-im2col + GEMM + ReLU, WMMA f16->f32.
// Block = 256 threads (8 wave32).  Output tile: 256 Cout x 32 pixels.
// K loop is split per deform group: the group's 9x32 bilinear table
// slice (9 KB) is staged into LDS once and reused for all 64 channels
// of the group (64x less table traffic).  Each wave owns 32x32 ->
// 4 accumulators, 4 v_wmma per K-step vs 8 ds_load_b128 fragment loads.
//   A stored row-major: As[m][k], lane reads K = gg*16 .. gg*16+15
//   B stored transposed: Bs[n][k], lane reads K = gg*16 .. gg*16+15
// =====================================================================
__global__ __launch_bounds__(256)
void dcn_gemm_kernel(const float* __restrict__ x,     // [B, C, H, W]
                     const float* __restrict__ wgt,   // [Cout, C*9] row-major
                     const Tab*   __restrict__ tab,   // [B, DG, 9, HW]
                     float* __restrict__ out)         // [B, Cout, H, W]
{
    __shared__ _Float16 As[M_TILE][LDS_STRIDE];   // weights (contiguous K order)
    __shared__ _Float16 Bs[N_TILE][LDS_STRIDE];   // sampled col, transposed
    __shared__ Tab      Et[KPTS][N_TILE];         // group's table slice (9 KB)

    const int b       = blockIdx.z;
    const int pixBase = blockIdx.x * N_TILE;
    const int tid  = threadIdx.x;
    const int lane = tid & 31;
    const int wid  = tid >> 5;

    v8f acc00 = {};   // (m-sub 0, n-sub 0)
    v8f acc01 = {};   // (m-sub 0, n-sub 1)
    v8f acc10 = {};   // (m-sub 1, n-sub 0)
    v8f acc11 = {};   // (m-sub 1, n-sub 1)

    // A loader: thread -> (row m_ld (+128), K half: 0..15 or 16..31)
    const int m_ld    = tid >> 1;                 // 0..127
    const int half_ld = (tid & 1) << 4;           // 0 or 16
    const int dst_lo  = (tid & 1) ? 16 : 0;       // first 8 halves (contiguous K)
    const int dst_hi  = (tid & 1) ? 24 : 8;       // second 8 halves (contiguous K)
    // B builder: thread -> (pixel n_b, 4 consecutive K values)
    const int n_b   = tid >> 3;                   // 0..31
    const int kk_b  = (tid & 7) << 2;             // 0,4,...,28

    // fragment read addresses (constant across K loop)
    const int gg   = lane >> 4;                   // lane group
    const int arow = (wid << 5) + (lane & 15);    // wave owns rows [wid*32, +32)
    const v16h* aptr0 = (const v16h*)&As[arow][gg << 4];
    const v16h* aptr1 = (const v16h*)&As[arow + 16][gg << 4];
    const v16h* b0ptr = (const v16h*)&Bs[lane & 15][gg << 4];
    const v16h* b1ptr = (const v16h*)&Bs[(lane & 15) + 16][gg << 4];

    for (int g = 0; g < DG; ++g) {
        // ---- stage this group's 9x32 table slice into LDS (coalesced) ----
        {
            const size_t base = (((size_t)b * DG + g) * KPTS) * HWN + pixBase;
            for (int t = tid; t < KPTS * N_TILE; t += 256) {
                const int k = t >> 5;             // kernel point 0..8
                const int n = t & 31;             // pixel in tile
                Et[k][n] = tab[base + (size_t)k * HWN + n];
            }
        }
        __syncthreads();

        for (int s = 0; s < GK / K_STEP; ++s) {   // 18 K-steps per group
            const int k0 = g * GK + s * K_STEP;
            // ---- stage weights -> LDS (f32 -> f16, contiguous K order) ----
            #pragma unroll
            for (int p = 0; p < 2; ++p) {
                const int row = m_ld + (p << 7);  // m_ld, m_ld + 128
                const float* wrow = wgt + (size_t)row * CK + k0 + half_ld;
                __builtin_prefetch(wrow + K_STEP, 0, 3);   // global_prefetch_b8
                const float4 v0 = *(const float4*)(wrow + 0);
                const float4 v1 = *(const float4*)(wrow + 4);
                const float4 v2 = *(const float4*)(wrow + 8);
                const float4 v3 = *(const float4*)(wrow + 12);
                h8 lo = { (_Float16)v0.x, (_Float16)v0.y, (_Float16)v0.z, (_Float16)v0.w,
                          (_Float16)v1.x, (_Float16)v1.y, (_Float16)v1.z, (_Float16)v1.w };
                h8 hi = { (_Float16)v2.x, (_Float16)v2.y, (_Float16)v2.z, (_Float16)v2.w,
                          (_Float16)v3.x, (_Float16)v3.y, (_Float16)v3.z, (_Float16)v3.w };
                *(h8*)&As[row][dst_lo] = lo;
                *(h8*)&As[row][dst_hi] = hi;
            }
            // ---- build sampled im2col tile -> LDS (table from LDS) ----
            {
                h4 col;
                #pragma unroll
                for (int j = 0; j < 4; ++j) {
                    const int ckl = s * K_STEP + kk_b + j;   // 0..575 in group
                    const int cg  = ckl / KPTS;
                    const int kp  = ckl - cg * KPTS;
                    const int c   = (g << 6) + cg;
                    const float* xc = x + ((size_t)b * CIN + c) * HWN;
                    const Tab e = Et[kp][n_b];
                    const float v = e.w.x * xc[e.idx.x] + e.w.y * xc[e.idx.y]
                                  + e.w.z * xc[e.idx.z] + e.w.w * xc[e.idx.w];
                    col[j] = (_Float16)v;
                }
                *(h4*)&Bs[n_b][kk_b] = col;
            }
            __syncthreads();

            // ---- contiguous 32B fragment loads + 4x WMMA ----
            const v16h a0 = *aptr0;
            const v16h a1 = *aptr1;
            const v16h b0 = *b0ptr;
            const v16h b1 = *b1ptr;

            acc00 = __builtin_amdgcn_wmma_f32_16x16x32_f16(
                        false, a0, false, b0, (short)0, acc00, false, false);
            acc01 = __builtin_amdgcn_wmma_f32_16x16x32_f16(
                        false, a0, false, b1, (short)0, acc01, false, false);
            acc10 = __builtin_amdgcn_wmma_f32_16x16x32_f16(
                        false, a1, false, b0, (short)0, acc10, false, false);
            acc11 = __builtin_amdgcn_wmma_f32_16x16x32_f16(
                        false, a1, false, b1, (short)0, acc11, false, false);
            __syncthreads();
        }
    }

    // ---- epilogue: ReLU + store (C/D layout: VGPR r -> M = gg*8 + r) ----
    const int n = lane & 15;
    #pragma unroll
    for (int r = 0; r < 8; ++r) {
        const int cout0 = (wid << 5) + (gg << 3) + r;        // m-sub 0
        float* o0 = out + ((size_t)b * COUT + cout0) * HWN + pixBase;
        o0[n]      = fmaxf(acc00[r], 0.f);
        o0[n + 16] = fmaxf(acc01[r], 0.f);
        float* o1 = o0 + (size_t)16 * HWN;                   // m-sub 1 (+16 rows)
        o1[n]      = fmaxf(acc10[r], 0.f);
        o1[n + 16] = fmaxf(acc11[r], 0.f);
    }
}

// =====================================================================
// Host launcher
// =====================================================================
extern "C" void kernel_launch(void* const* d_in, const int* in_sizes, int n_in,
                              void* d_out, int out_size, void* d_ws, size_t ws_size,
                              hipStream_t stream) {
    const float* reg_feat = (const float*)d_in[0];   // [8,256,64,64]
    const float* cls_feat = (const float*)d_in[1];   // [8,256,64,64]
    const float* pred     = (const float*)d_in[2];   // [8,34,64,64]
    const float* w_off    = (const float*)d_in[3];   // [72,34]
    const float* w_reg    = (const float*)d_in[4];   // [256,256,3,3]
    const float* w_cls    = (const float*)d_in[5];   // [256,256,3,3]
    float* out = (float*)d_out;                      // 2 x [8,256,64,64] concat
    Tab* tab = (Tab*)d_ws;                           // needs B*DG*9*HW*32B = 36 MB

    (void)in_sizes; (void)n_in; (void)out_size; (void)ws_size;

    // 1) shared bilinear table (offset conv folded in)
    {
        const int nt = BATCH * DG * KPTS * HWN;
        dcn_precomp_kernel<<<dim3((nt + 255) / 256), dim3(256), 0, stream>>>(
            pred, w_off, tab);
    }
    // 2) two fused deformable GEMMs (share the table)
    {
        dim3 grid(HWN / N_TILE, 1, BATCH);   // 128 x 1 x 8 per conv
        dim3 block(256);
        dcn_gemm_kernel<<<grid, block, 0, stream>>>(reg_feat, w_reg, tab, out);
        dcn_gemm_kernel<<<grid, block, 0, stream>>>(
            cls_feat, w_cls, tab, out + (size_t)BATCH * COUT * HWN);
    }
}
